// simpleLSTM_37374805410597
// MI455X (gfx1250) — matrix operand order repaired and checked by reading, not compiled, not run on hardware
//
#include <hip/hip_runtime.h>
#include <stdint.h>
#include <stddef.h>

// Problem constants (match reference)
#define B_    16
#define T_    128
#define HD    1024
#define FOURH 4096
#define VOC   32000
#define MROWS (B_ * T_)        // 2048 rows, row = t*16 + b
#define KSTEPS (HD / 32)       // 32

typedef __bf16 bf16;
typedef __attribute__((ext_vector_type(16))) __bf16 v16bf;
typedef __attribute__((ext_vector_type(8)))  __bf16 v8bf;
typedef __attribute__((ext_vector_type(8)))  float  v8f;
typedef int v4i __attribute__((vector_size(16)));   // matches builtin param type

// CDNA5 async global->LDS path (guarded; falls back to register staging)
#if defined(__has_builtin)
#  if __has_builtin(__builtin_amdgcn_global_load_async_to_lds_b128) && \
      __has_builtin(__builtin_amdgcn_s_wait_asynccnt)
#    define USE_ASYNC_LDS 1
#  endif
#endif
#ifndef USE_ASYNC_LDS
#  define USE_ASYNC_LDS 0
#endif

#define AS1 __attribute__((address_space(1)))
#define AS3 __attribute__((address_space(3)))

// ---------------------------------------------------------------------------
// WMMA fragment loaders (contiguous-k layouts used by this code, K stride = HD = 1024)
// A: 16x32 bf16, lane m=lane&15, half=lane>>4; k = half*16 + e  (16 contiguous bf16)
// B: 32x16 bf16, lane n=lane&15; k = half*16 + e  (16 contiguous bf16)
// ---------------------------------------------------------------------------
static __device__ __forceinline__ v16bf frag_a(const bf16* __restrict__ base,
                                               int kbase, int lane) {
  const int m = lane & 15, half = lane >> 4;
  const bf16* p = base + (size_t)m * HD + kbase + half * 16;
  v8bf lo = *(const v8bf*)(p);
  v8bf hi = *(const v8bf*)(p + 8);
  return __builtin_shufflevector(lo, hi, 0,1,2,3,4,5,6,7,8,9,10,11,12,13,14,15);
}
static __device__ __forceinline__ v16bf frag_b(const bf16* __restrict__ base,
                                               int kbase, int lane) {
  const int n = lane & 15, half = lane >> 4;
  const bf16* p = base + (size_t)n * HD + kbase + half * 16;
  v8bf lo = *(const v8bf*)(p);
  v8bf hi = *(const v8bf*)(p + 8);
  return __builtin_shufflevector(lo, hi, 0,1,2,3,4,5,6,7,8,9,10,11,12,13,14,15);
}
#define WMMA_BF16(a, b, c) \
  __builtin_amdgcn_wmma_f32_16x16x32_bf16(false, (a), false, (b), (short)0, (c), false, false)

static __device__ __forceinline__ float sigmoidf_(float x) {
  return 1.0f / (1.0f + __expf(-x));
}

// ---------------------------------------------------------------------------
// Shared GEMM core: 256 threads (8 waves). Wave w computes one 16-row M-tile
// against 8 N-tiles (n0..n0+127). B tiles staged in LDS in fragment layout,
// double-buffered. LDS slot: [buf][tile][lane] -> 24 bf16 (32B + 16B pad).
// ---------------------------------------------------------------------------
#define LDS_ELEMS (2 * 8 * 32 * 24)
#define BUFSTRIDE (8 * 32 * 24)

static __device__ __forceinline__ void gemm_core(const bf16* __restrict__ Abase,
                                                 const bf16* __restrict__ Bn0,
                                                 bf16* __restrict__ lds,
                                                 v8f (&acc)[8]) {
  const int tid  = threadIdx.x;
  const int lane = tid & 31;
  // copy role: thread tid owns slot (tile = tid>>5, slot-lane = tid&31): 32B
  const bf16* gB = Bn0 + (size_t)(((tid >> 5) << 4) + (tid & 15)) * HD
                       + (((tid >> 4) & 1) << 4);
  bf16* myslot = lds + (size_t)(tid) * 24;   // (tid>>5)*32 + (tid&31) == tid

#if USE_ASYNC_LDS
  // ---- CDNA5 async DMA staging: global_load_async_to_lds_b128 (ASYNCcnt) ----
  auto issue_async = [&](int kk2) {
    const bf16* src = gB + kk2 * 32;
    bf16* dst = myslot + (kk2 & 1) * BUFSTRIDE;
    // imm offset is added to BOTH global and LDS addresses (ISA 15.18.3)
    __builtin_amdgcn_global_load_async_to_lds_b128(
        (AS1 v4i*)(uintptr_t)src, (AS3 v4i*)(unsigned)(uintptr_t)dst, 0, 0);
    __builtin_amdgcn_global_load_async_to_lds_b128(
        (AS1 v4i*)(uintptr_t)src, (AS3 v4i*)(unsigned)(uintptr_t)dst, 16, 0);
  };
  issue_async(0);                       // prime both buffers: 4 outstanding
  issue_async(1);
  v16bf aCur = frag_a(Abase, 0, lane);

  for (int kk = 0; kk < KSTEPS; ++kk) {
    if (kk + 1 < KSTEPS) __builtin_amdgcn_s_wait_asynccnt(2);  // buf(kk) ready
    else                 __builtin_amdgcn_s_wait_asynccnt(0);
    __syncthreads();                    // all waves' copies for kk complete

    v16bf aNext = aCur;
    if (kk + 1 < KSTEPS) {
      aNext = frag_a(Abase, (kk + 1) * 32, lane);
      __builtin_prefetch(gB + (kk + 2) * 32, 0, 1);   // CDNA5 global_prefetch
    }
    const bf16* rbuf = lds + (kk & 1) * BUFSTRIDE;
#pragma unroll
    for (int j = 0; j < 8; ++j) {
      const bf16* p = rbuf + (size_t)(j * 32 + lane) * 24;
      v8bf lo = *(const v8bf*)(p);
      v8bf hi = *(const v8bf*)(p + 8);
      v16bf b = __builtin_shufflevector(lo, hi,
                    0,1,2,3,4,5,6,7,8,9,10,11,12,13,14,15);
      acc[j] = WMMA_BF16(aCur, b, acc[j]);
    }
    aCur = aNext;
    __syncthreads();                    // everyone done reading buf(kk)
    if (kk + 2 < KSTEPS) issue_async(kk + 2);   // refill freed buffer
  }
#else
  // ---- fallback: register-staged copy (global -> VGPR -> ds_store) ----
  v8bf r0 = *(const v8bf*)(gB);
  v8bf r1 = *(const v8bf*)(gB + 8);
  v16bf aCur = frag_a(Abase, 0, lane);

  for (int kk = 0; kk < KSTEPS; ++kk) {
    bf16* dst = myslot + (kk & 1) * BUFSTRIDE;
    *(v8bf*)(dst)     = r0;      // waits gB(kk); hidden behind wmma(kk-1)
    *(v8bf*)(dst + 8) = r1;
    __syncthreads();
    if (kk + 1 < KSTEPS) {       // issue global loads for k+1 (not waited here)
      const bf16* g = gB + (kk + 1) * 32;
      r0 = *(const v8bf*)(g);
      r1 = *(const v8bf*)(g + 8);
      __builtin_prefetch(gB + (kk + 2) * 32, 0, 1);
    }
    v16bf aNext = aCur;
    if (kk + 1 < KSTEPS) aNext = frag_a(Abase, (kk + 1) * 32, lane);

    const bf16* rbuf = lds + (kk & 1) * BUFSTRIDE;
#pragma unroll
    for (int j = 0; j < 8; ++j) {
      const bf16* p = rbuf + (size_t)(j * 32 + lane) * 24;
      v8bf lo = *(const v8bf*)(p);
      v8bf hi = *(const v8bf*)(p + 8);
      v16bf b = __builtin_shufflevector(lo, hi,
                    0,1,2,3,4,5,6,7,8,9,10,11,12,13,14,15);
      acc[j] = WMMA_BF16(aCur, b, acc[j]);
    }
    aCur = aNext;
  }
#endif
}

// ---------------------------------------------------------------------------
// 1) f32 -> bf16 conversion (weights)
// ---------------------------------------------------------------------------
__global__ void cvt_bf16_kernel(const float* __restrict__ in, bf16* __restrict__ out, int n) {
  int i = blockIdx.x * blockDim.x + threadIdx.x;
  int stride = gridDim.x * blockDim.x;
  for (; i < n; i += stride) out[i] = (bf16)in[i];
}

// ---------------------------------------------------------------------------
// 2) Embedding gather -> Xbf [row = t*16 + b][HD] bf16
// ---------------------------------------------------------------------------
__global__ void embed_kernel(const int* __restrict__ idx, const float* __restrict__ Wemb,
                             bf16* __restrict__ Xbf) {
  int e = blockIdx.x * blockDim.x + threadIdx.x;
  if (e >= MROWS * HD) return;
  int col = e & (HD - 1);
  int row = e >> 10;          // t*16 + b
  int t = row >> 4, b = row & 15;
  int w = idx[b * T_ + t];
  Xbf[e] = (bf16)Wemb[(size_t)w * HD + col];
}

// ---------------------------------------------------------------------------
// 3) Xp = Xbf @ Wx2^T + b2   ([2048 x 4096] f32)   grid(32, 16), block 256
// ---------------------------------------------------------------------------
__global__ __launch_bounds__(256) void gemm_xp_kernel(const bf16* __restrict__ Xbf,
                                                      const bf16* __restrict__ Wxbf,
                                                      const float* __restrict__ b2,
                                                      float* __restrict__ Xp) {
  __shared__ __align__(16) bf16 lds[LDS_ELEMS];
  const int lane = threadIdx.x & 31;
  const int wave = threadIdx.x >> 5;
  const int mtile = blockIdx.y * 8 + wave;       // 0..127
  const int n0 = blockIdx.x * 128;

  v8f acc[8] = {};
  gemm_core(Xbf + (size_t)mtile * 16 * HD, Wxbf + (size_t)n0 * HD, lds, acc);

  const int n_in = lane & 15, half = lane >> 4;
#pragma unroll
  for (int j = 0; j < 8; ++j) {
    int v = n0 + j * 16 + n_in;
    float bias = b2[v];
#pragma unroll
    for (int r = 0; r < 8; ++r) {
      int m = r + 8 * half;
      Xp[(size_t)(mtile * 16 + m) * FOURH + v] = acc[j][r] + bias;
    }
  }
}

// ---------------------------------------------------------------------------
// 4) Persistent LSTM recurrence. 16 WGs x 4 waves = 64 waves; wave owns one
//    16-unit tile (all 4 gates). Device-wide generation barrier between steps.
//    Hbf[t] = h emitted BEFORE step t (reference quirk) == step-t GEMM input.
//    All 5 fragments register double-buffered to hide L2 latency.
// ---------------------------------------------------------------------------
static __device__ __forceinline__ void grid_barrier(unsigned* bar, unsigned* gen,
                                                    unsigned nwg) {
  __threadfence();          // make this wave's global stores device-visible
  __syncthreads();
  if (threadIdx.x == 0) {
    unsigned g = __hip_atomic_load(gen, __ATOMIC_ACQUIRE, __HIP_MEMORY_SCOPE_AGENT);
    unsigned a = __hip_atomic_fetch_add(bar, 1u, __ATOMIC_ACQ_REL, __HIP_MEMORY_SCOPE_AGENT);
    if (a == nwg - 1) {
      __hip_atomic_store(bar, 0u, __ATOMIC_RELAXED, __HIP_MEMORY_SCOPE_AGENT);
      __hip_atomic_fetch_add(gen, 1u, __ATOMIC_RELEASE, __HIP_MEMORY_SCOPE_AGENT);
    } else {
      while (__hip_atomic_load(gen, __ATOMIC_ACQUIRE, __HIP_MEMORY_SCOPE_AGENT) == g)
        __builtin_amdgcn_s_sleep(1);
    }
  }
  __syncthreads();
}

__global__ __launch_bounds__(128) void lstm_kernel(const bf16* __restrict__ Whbf,
                                                   const float* __restrict__ Xp,
                                                   bf16* __restrict__ Hbf,
                                                   float* __restrict__ cbuf,
                                                   unsigned* __restrict__ bar) {
  const int lane = threadIdx.x & 31;
  const int wave = threadIdx.x >> 5;
  const int u = blockIdx.x * 4 + wave;   // hidden-unit tile 0..63
  const int j0 = u * 16;
  const int n_in = lane & 15, half = lane >> 4;
  unsigned* gen = bar + 64;              // second counter in 512B region
  const bf16* WhF = Whbf + (size_t)(0 * HD + j0) * HD;
  const bf16* WhG = Whbf + (size_t)(1 * HD + j0) * HD;
  const bf16* WhI = Whbf + (size_t)(2 * HD + j0) * HD;
  const bf16* WhO = Whbf + (size_t)(3 * HD + j0) * HD;

  for (int t = 0; t < T_; ++t) {
    const bf16* Abase = Hbf + (size_t)t * B_ * HD;
    v8f af_ = {}, ag_ = {}, ai_ = {}, ao_ = {};

    v16bf a0  = frag_a(Abase, 0, lane);
    v16bf bf0 = frag_b(WhF, 0, lane);
    v16bf bg0 = frag_b(WhG, 0, lane);
    v16bf bi0 = frag_b(WhI, 0, lane);
    v16bf bo0 = frag_b(WhO, 0, lane);
    for (int k = 0; k < HD; k += 32) {
      v16bf a1 = a0, bf1 = bf0, bg1 = bg0, bi1 = bi0, bo1 = bo0;
      if (k + 32 < HD) {                 // issue next-k loads before the MACs
        a1  = frag_a(Abase, k + 32, lane);
        bf1 = frag_b(WhF, k + 32, lane);
        bg1 = frag_b(WhG, k + 32, lane);
        bi1 = frag_b(WhI, k + 32, lane);
        bo1 = frag_b(WhO, k + 32, lane);
      }
      af_ = WMMA_BF16(a0, bf0, af_);
      ag_ = WMMA_BF16(a0, bg0, ag_);
      ai_ = WMMA_BF16(a0, bi0, ai_);
      ao_ = WMMA_BF16(a0, bo0, ao_);
      a0 = a1; bf0 = bf1; bg0 = bg1; bi0 = bi1; bo0 = bo1;
    }
#pragma unroll
    for (int r = 0; r < 8; ++r) {
      int m = r + 8 * half;              // batch index
      int j = j0 + n_in;                 // hidden unit
      const float* xp = Xp + (size_t)(t * B_ + m) * FOURH;
      float pf = tanhf(af_[r] + xp[0 * HD + j]);   // A = tanh(pre) quirk
      float pg = tanhf(ag_[r] + xp[1 * HD + j]);
      float pi = tanhf(ai_[r] + xp[2 * HD + j]);
      float po = tanhf(ao_[r] + xp[3 * HD + j]);
      float fg = sigmoidf_(pf);
      float gg = tanhf(pg);
      float ig = sigmoidf_(pi);
      float og = sigmoidf_(po);
      float cold = cbuf[m * HD + j];
      float cnew = fg * cold + gg * ig;
      float hnew = og * tanhf(cold);               // PRE-update cell (quirk)
      cbuf[m * HD + j] = cnew;
      if (t + 1 < T_)
        Hbf[(size_t)(t + 1) * B_ * HD + (size_t)m * HD + j] = (bf16)hnew;
    }
    grid_barrier(bar, gen, gridDim.x);
  }
}

// ---------------------------------------------------------------------------
// 5) Logits = Hbf @ Wa^T + ba, stored permuted into out[b][t][v] (f32).
//    M-tile index == t. grid(250, 16), block 256.
// ---------------------------------------------------------------------------
__global__ __launch_bounds__(256) void gemm_out_kernel(const bf16* __restrict__ Hbf,
                                                       const bf16* __restrict__ Wabf,
                                                       const float* __restrict__ ba,
                                                       float* __restrict__ out) {
  __shared__ __align__(16) bf16 lds[LDS_ELEMS];
  const int lane = threadIdx.x & 31;
  const int wave = threadIdx.x >> 5;
  const int mtile = blockIdx.y * 8 + wave;       // == t, 0..127
  const int n0 = blockIdx.x * 128;

  v8f acc[8] = {};
  gemm_core(Hbf + (size_t)mtile * 16 * HD, Wabf + (size_t)n0 * HD, lds, acc);

  const int n_in = lane & 15, half = lane >> 4;
  const int t = mtile;
#pragma unroll
  for (int j = 0; j < 8; ++j) {
    int v = n0 + j * 16 + n_in;
    float bias = ba[v];
#pragma unroll
    for (int r = 0; r < 8; ++r) {
      int m = r + 8 * half;              // batch index
      out[((size_t)m * T_ + t) * VOC + v] = acc[j][r] + bias;
    }
  }
}

// ---------------------------------------------------------------------------
// 6) softmax over axis 0 (the 16 batches) per (t, v) — reference quirk
// ---------------------------------------------------------------------------
__global__ void softmax_b_kernel(float* __restrict__ out) {
  size_t p = (size_t)blockIdx.x * blockDim.x + threadIdx.x;   // t*VOC + v
  const size_t NP = (size_t)T_ * VOC;                         // 4,096,000
  if (p >= NP) return;
  float v[B_];
  float mx = -3.402823466e+38f;
#pragma unroll
  for (int b = 0; b < B_; ++b) { v[b] = out[p + (size_t)b * NP]; mx = fmaxf(mx, v[b]); }
  float s = 0.f;
#pragma unroll
  for (int b = 0; b < B_; ++b) { v[b] = __expf(v[b] - mx); s += v[b]; }
  float inv = 1.f / s;
#pragma unroll
  for (int b = 0; b < B_; ++b) out[p + (size_t)b * NP] = v[b] * inv;
}

// ---------------------------------------------------------------------------
// Host launcher
// ---------------------------------------------------------------------------
extern "C" void kernel_launch(void* const* d_in, const int* in_sizes, int n_in,
                              void* d_out, int out_size, void* d_ws, size_t ws_size,
                              hipStream_t stream) {
  const int*   idx   = (const int*)  d_in[0];
  const float* Wemb  = (const float*)d_in[1];
  const float* Wx2   = (const float*)d_in[2];
  const float* Wh2   = (const float*)d_in[3];
  const float* b2    = (const float*)d_in[4];
  const float* Wa    = (const float*)d_in[5];
  const float* ba    = (const float*)d_in[6];
  float* out = (float*)d_out;

  // workspace carve (256B aligned)
  char* base = (char*)d_ws;
  size_t off = 0;
  auto carve = [&](size_t bytes) -> char* {
    off = (off + 255) & ~(size_t)255;
    char* p = base + off;
    off += bytes;
    return p;
  };
  bf16*     Xbf  = (bf16*)    carve((size_t)MROWS * HD * 2);     //   4 MiB
  bf16*     Wxbf = (bf16*)    carve((size_t)FOURH * HD * 2);     //   8 MiB
  bf16*     Whbf = (bf16*)    carve((size_t)FOURH * HD * 2);     //   8 MiB
  bf16*     Wabf = (bf16*)    carve((size_t)VOC * HD * 2);       //  ~63 MiB
  float*    Xp   = (float*)   carve((size_t)MROWS * FOURH * 4);  //  32 MiB
  bf16*     Hbf  = (bf16*)    carve((size_t)T_ * B_ * HD * 2);   //   4 MiB
  float*    cbuf = (float*)   carve((size_t)B_ * HD * 4);        //  64 KiB
  unsigned* bar  = (unsigned*)carve(512);                        // [0]=cnt [64]=gen
  (void)ws_size; (void)n_in; (void)in_sizes; (void)out_size;

  // deterministic per-call init (graph-capture legal)
  (void)hipMemsetAsync(Hbf,  0, (size_t)B_ * HD * 2, stream);   // Hbf[0] = h0 = 0
  (void)hipMemsetAsync(cbuf, 0, (size_t)B_ * HD * 4, stream);   // c0 = 0
  (void)hipMemsetAsync(bar,  0, 512, stream);                   // barrier counters

  // weight conversions to bf16
  {
    int n;
    n = FOURH * HD; cvt_bf16_kernel<<<4096, 256, 0, stream>>>(Wx2, Wxbf, n);
    n = FOURH * HD; cvt_bf16_kernel<<<4096, 256, 0, stream>>>(Wh2, Whbf, n);
    n = VOC * HD;   cvt_bf16_kernel<<<8192, 256, 0, stream>>>(Wa,  Wabf, n);
  }

  // embedding gather
  embed_kernel<<<(MROWS * HD) / 256, 256, 0, stream>>>(idx, Wemb, Xbf);

  // Xp = X @ Wx2^T + b2
  gemm_xp_kernel<<<dim3(FOURH / 128, MROWS / 128), 256, 0, stream>>>(Xbf, Wxbf, b2, Xp);

  // sequential recurrence (persistent, device-wide barrier per step)
  lstm_kernel<<<16, 128, 0, stream>>>(Whbf, Xp, Hbf, cbuf, bar);

  // logits = H @ Wa^T + ba  (permuted store into out[b][t][v])
  gemm_out_kernel<<<dim3(VOC / 128, MROWS / 128), 256, 0, stream>>>(Hbf, Wabf, ba, out);

  // softmax over batch axis
  softmax_b_kernel<<<((T_ * VOC) + 255) / 256, 256, 0, stream>>>(out);
}
